// LiquidLayer_59648505807317
// MI455X (gfx1250) — matrix operand order repaired and checked by reading, not compiled, not run on hardware
//
#include <hip/hip_runtime.h>

// Problem dimensions (from the reference)
#define BATCH  64
#define STEPS  1024
#define INPUT  512
#define HIDDEN 1024

typedef __bf16 bf16x16 __attribute__((ext_vector_type(16)));
typedef float  f32x4   __attribute__((ext_vector_type(4)));
typedef float  f32x8   __attribute__((ext_vector_type(8)));
typedef int    v4i     __attribute__((ext_vector_type(4)));

#if __has_builtin(__builtin_amdgcn_global_load_async_to_lds_b128) && \
    __has_builtin(__builtin_amdgcn_s_wait_asynccnt)
#define ASYNC_LDS_COPY 1
// Builtin expects address-space-qualified int4 pointers (per hipcc diagnostic):
typedef __attribute__((address_space(1))) v4i* as1_v4i_p;  // global
typedef __attribute__((address_space(3))) v4i* as3_v4i_p;  // LDS
#endif

// ---------------------------------------------------------------------------
// K-permutation (same as before): A operand is one contiguous 32B load per
// lane (lane-half h holds logical K [kb+16h, kb+16h+15]); weights are
// pre-shuffled so the matching B operand is also one contiguous 32B load:
//   Wp[(((n*(K/32)+kb32)*2)+h)*16 + j] = bf16(W[n*K + kb32*32 + koff]),
//   koff = 16h+j
// ---------------------------------------------------------------------------
__global__ void pack_w_bf16(const float* __restrict__ W,
                            unsigned short* __restrict__ Wp,
                            int K, int total /* = H*K */) {
    int tid = blockIdx.x * blockDim.x + threadIdx.x;
    if (tid >= total) return;
    const int j    = tid & 15;
    const int h    = (tid >> 4) & 1;
    const int blk  = tid >> 5;            // n*(K/32) + kb32
    const int kb32 = blk & ((K >> 5) - 1);
    const int n    = blk >> (K == 512 ? 4 : 5);
    const int koff = h * 16 + j;
    const float v  = W[(size_t)n * K + kb32 * 32 + koff];
    Wp[tid] = __builtin_bit_cast(unsigned short, (__bf16)v);
}

// Zero h0 exchange buffer and the grid-barrier counters
__global__ void init_state(unsigned short* __restrict__ hb0,
                           unsigned* __restrict__ counters) {
    int i = blockIdx.x * blockDim.x + threadIdx.x;
    if (i < BATCH * HIDDEN) hb0[i] = 0;
    if (i < STEPS) counters[i] = 0;
}

// ---------------------------------------------------------------------------
// Phase 1: u_in[b*T+t, n] = sum_i x[b,t,i]*W_in[n,i]  -> written into d_out
// grid = (B*T/16, H/128), block = 256 (8 waves; wave w owns N-tile by*8+w)
// ---------------------------------------------------------------------------
__global__ void proj_kernel(const float* __restrict__ x,
                            const unsigned short* __restrict__ win_p,
                            float* __restrict__ u) {
    const int lane = threadIdx.x & 31;
    const int wave = threadIdx.x >> 5;
    const int half = lane >> 4;
    const int lidx = lane & 15;
    const int m0   = blockIdx.x * 16;
    const int n0   = (blockIdx.y * 8 + wave) * 16;
    const int ncol = n0 + lidx;

    const float*          ap = x + (size_t)(m0 + lidx) * INPUT + half * 16;
    const unsigned short* bp = win_p + (size_t)ncol * INPUT + half * 16;

    f32x8 acc = {};
    #pragma unroll 4
    for (int kb = 0; kb < INPUT; kb += 32) {
        f32x4 a0 = *(const f32x4*)(ap + kb);
        f32x4 a1 = *(const f32x4*)(ap + kb + 4);
        f32x4 a2 = *(const f32x4*)(ap + kb + 8);
        f32x4 a3 = *(const f32x4*)(ap + kb + 12);
        bf16x16 A;
        #pragma unroll
        for (int i = 0; i < 4; ++i) {
            A[i]      = (__bf16)a0[i];
            A[4 + i]  = (__bf16)a1[i];
            A[8 + i]  = (__bf16)a2[i];
            A[12 + i] = (__bf16)a3[i];
        }
        bf16x16 B = *(const bf16x16*)(bp + kb);
        acc = __builtin_amdgcn_wmma_f32_16x16x32_bf16(false, A, false, B,
                                                      (short)0, acc, false, false);
    }
    #pragma unroll
    for (int r = 0; r < 8; ++r) {
        const int mr = m0 + r + half * 8;
        u[(size_t)mr * HIDDEN + ncol] = acc[r];
    }
}

// ---------------------------------------------------------------------------
// Device-wide barrier: one arrival per block, agent-scope release/acquire.
// Counter array has one slot per step (zeroed each launch) -> no reset races.
// ---------------------------------------------------------------------------
__device__ __forceinline__ void grid_barrier(unsigned* counters, int t,
                                             unsigned nblocks) {
    __syncthreads();
    if (threadIdx.x == 0) {
        __hip_atomic_fetch_add(&counters[t], 1u, __ATOMIC_RELEASE,
                               __HIP_MEMORY_SCOPE_AGENT);
        while (__hip_atomic_load(&counters[t], __ATOMIC_ACQUIRE,
                                 __HIP_MEMORY_SCOPE_AGENT) < nblocks) {
            __builtin_amdgcn_s_sleep(1);
        }
    }
    __syncthreads();
}

// ---------------------------------------------------------------------------
// Phase 2: persistent scan over all 1024 steps.
//   u  = u_in(t) + h @ W_rec^T ;  h' = h + (-h + alpha*tanh(u+beta))/tau
// grid = 64 blocks (one per 16-col N-tile) x 128 threads (4 waves = 4 M-tiles).
// W_rec tile (32 KB) staged in LDS once; fp32 h state lives in registers for
// the whole scan; only the bf16 exchange copy round-trips through L2.
// ---------------------------------------------------------------------------
__global__ void __launch_bounds__(128)
scan_kernel(const unsigned short* __restrict__ wrec_p,
            const float* __restrict__ tau,
            const float* __restrict__ alpha,
            const float* __restrict__ beta,
            float* __restrict__ out,
            unsigned short* __restrict__ hb0,
            unsigned short* __restrict__ hb1,
            unsigned* __restrict__ counters) {
    // LDS layout: chunk index g = kb32*2 + half; address (u16) = (g*16+col)*16+j
    __shared__ unsigned short bsm[16 * HIDDEN];   // 32 KB

    const int tid  = threadIdx.x;
    const int lane = tid & 31;
    const int wave = tid >> 5;
    const int half = lane >> 4;
    const int lidx = lane & 15;
    const int n0   = blockIdx.x * 16;
    const int m0   = wave * 16;
    const int ncol = n0 + lidx;

    // ---- stage packed W_rec tile into LDS (transposed: col-major -> chunk-major)
    #pragma unroll 4
    for (int it = 0; it < 16; ++it) {
        const int c   = it * 128 + tid;   // 16B piece id, 0..2047
        const int col = c & 15;
        const int p   = c >> 4;           // 0..127 within a column's packed stream
        const unsigned short* src = wrec_p + (size_t)(n0 + col) * HIDDEN + p * 8;
        unsigned short*       dst = &bsm[(((p >> 1) * 16 + col) * 16) + (p & 1) * 8];
#ifdef ASYNC_LDS_COPY
        __builtin_amdgcn_global_load_async_to_lds_b128((as1_v4i_p)src,
                                                       (as3_v4i_p)dst, 0, 0);
#else
        *(uint4*)dst = *(const uint4*)src;
#endif
    }
#ifdef ASYNC_LDS_COPY
    __builtin_amdgcn_s_wait_asynccnt(0);
#endif
    __syncthreads();

    // ---- per-wave constants (fixed tile ownership across time)
    const float tv = tau[ncol];
    const float av = alpha[ncol];
    const float bv = beta[ncol];
    const float inv_tau = 1.0f / tv;

    float hreg[8];                         // fp32 state, register-resident
    #pragma unroll
    for (int r = 0; r < 8; ++r) hreg[r] = 0.0f;

    const unsigned nblocks = gridDim.x;

    for (int t = 0; t < STEPS; ++t) {
        const unsigned short* hb_in  = (t & 1) ? hb1 : hb0;
        unsigned short*       hb_out = (t & 1) ? hb0 : hb1;
        const unsigned short* ap = hb_in + (size_t)(m0 + lidx) * HIDDEN + half * 16;
        const unsigned short* bl = &bsm[(half * 16 + lidx) * 16];

        f32x8 acc = {};
        #pragma unroll 8
        for (int kb32 = 0; kb32 < HIDDEN / 32; ++kb32) {
            bf16x16 A = *(const bf16x16*)(ap + kb32 * 32);
            bf16x16 B = *(const bf16x16*)(bl + kb32 * 512);  // 2 chunks * 16 lanes * 16
            acc = __builtin_amdgcn_wmma_f32_16x16x32_bf16(false, A, false, B,
                                                          (short)0, acc, false, false);
        }

        #pragma unroll
        for (int r = 0; r < 8; ++r) {
            const int    m    = m0 + r + half * 8;
            const size_t oidx = ((size_t)m * STEPS + t) * HIDDEN + ncol; // (b,t,n)
            const float u  = acc[r] + out[oidx];          // add precomputed u_in
            const float hn = hreg[r] + (av * tanhf(u + bv) - hreg[r]) * inv_tau;
            out[oidx] = hn;                               // y_t = h_new
            hreg[r]   = hn;
            hb_out[(size_t)m * HIDDEN + ncol] =
                __builtin_bit_cast(unsigned short, (__bf16)hn);
        }

        grid_barrier(counters, t, nblocks);               // publish h(t+1)
    }
}

// ---------------------------------------------------------------------------
extern "C" void kernel_launch(void* const* d_in, const int* in_sizes, int n_in,
                              void* d_out, int out_size, void* d_ws, size_t ws_size,
                              hipStream_t stream) {
    const float* x     = (const float*)d_in[0];
    const float* W_in  = (const float*)d_in[1];
    const float* W_rec = (const float*)d_in[2];
    const float* tau   = (const float*)d_in[3];
    const float* alpha = (const float*)d_in[4];
    const float* beta  = (const float*)d_in[5];
    float* out = (float*)d_out;

    // Workspace layout (~3.4 MB):
    unsigned short* wrec_p = (unsigned short*)d_ws;              // H*H bf16, packed
    unsigned short* win_p  = wrec_p + (size_t)HIDDEN * HIDDEN;   // H*I bf16, packed
    unsigned short* hb0    = win_p + (size_t)HIDDEN * INPUT;     // B*H bf16 exchange
    unsigned short* hb1    = hb0 + BATCH * HIDDEN;
    unsigned*       cnt    = (unsigned*)(hb1 + BATCH * HIDDEN);  // STEPS counters

    pack_w_bf16<<<(HIDDEN * HIDDEN) / 256, 256, 0, stream>>>(W_rec, wrec_p,
                                                             HIDDEN, HIDDEN * HIDDEN);
    pack_w_bf16<<<(HIDDEN * INPUT) / 256, 256, 0, stream>>>(W_in, win_p,
                                                            INPUT, HIDDEN * INPUT);
    init_state<<<(BATCH * HIDDEN) / 256, 256, 0, stream>>>(hb0, cnt);

    // Phase 1: big input-projection GEMM into d_out
    proj_kernel<<<dim3((BATCH * STEPS) / 16, HIDDEN / 128), 256, 0, stream>>>(x, win_p, out);

    // Phase 2: persistent scan, single launch
    scan_kernel<<<dim3(HIDDEN / 16), 128, 0, stream>>>(wrec_p, tau, alpha, beta,
                                                       out, hb0, hb1, cnt);

    (void)in_sizes; (void)n_in; (void)out_size; (void)ws_size;
}
